// EdgePolicy_58497454571689
// MI455X (gfx1250) — matrix operand order repaired and checked by reading, not compiled, not run on hardware
//
#include <hip/hip_runtime.h>
#include <hip/hip_bf16.h>

typedef _Float16 f16;
typedef __attribute__((ext_vector_type(16))) _Float16 v16h;
typedef __attribute__((ext_vector_type(8)))  _Float16 v8h;
typedef __attribute__((ext_vector_type(8)))  float    v8f;
typedef __attribute__((ext_vector_type(8)))  unsigned short v8u;

#define HID 64
#define DIN 4
#define E3  192       // 3*HID
#define KPAD 200      // padded row stride in f16 (400B, 16B aligned, spreads banks)
#define WAVES 2       // waves per block in edge kernel (block = 64 threads)
#define MTILE 32      // edges per wave per iteration (two 16-row A tiles)

// ---------------------------------------------------------------------------
// Kernel 0: one-off transpose We1 (192x64 f32, row-major) -> WeT_g (64x192 f16)
// ---------------------------------------------------------------------------
__global__ void we1_transpose_kernel(const float* __restrict__ We1,
                                     f16* __restrict__ WeT_g) {
    const int i = blockIdx.x * blockDim.x + threadIdx.x;   // over 192*64
    if (i < E3 * HID) {
        const int k = i / HID, n = i % HID;
        WeT_g[n * E3 + k] = (f16)We1[i];
    }
}

// ---------------------------------------------------------------------------
// Kernel 1: node MLP  H = relu(relu(X @ W1 + b1) @ W2 + b2), stored as f16
// block = 256 threads = 4 nodes x 64 hidden units
// ---------------------------------------------------------------------------
__global__ void node_mlp_kernel(const float* __restrict__ X,
                                const float* __restrict__ W1,
                                const float* __restrict__ b1,
                                const float* __restrict__ W2,
                                const float* __restrict__ b2,
                                f16* __restrict__ H, int nNodes) {
    __shared__ float sh[4][HID];
    const int h    = threadIdx.x & (HID - 1);
    const int ln   = threadIdx.x >> 6;            // 0..3 local node
    const int node = blockIdx.x * 4 + ln;

    float a = 0.0f;
    if (node < nNodes) {
        const float x0 = X[node * DIN + 0];
        const float x1 = X[node * DIN + 1];
        const float x2 = X[node * DIN + 2];
        const float x3 = X[node * DIN + 3];
        a = b1[h];
        a = fmaf(x0, W1[0 * HID + h], a);
        a = fmaf(x1, W1[1 * HID + h], a);
        a = fmaf(x2, W1[2 * HID + h], a);
        a = fmaf(x3, W1[3 * HID + h], a);
        a = fmaxf(a, 0.0f);
    }
    sh[ln][h] = a;
    __syncthreads();
    if (node < nNodes) {
        float acc = b2[h];
#pragma unroll 16
        for (int j = 0; j < HID; ++j)
            acc = fmaf(sh[ln][j], W2[j * HID + h], acc);
        acc = fmaxf(acc, 0.0f);
        H[(size_t)node * HID + h] = (f16)acc;
    }
}

// ---------------------------------------------------------------------------
// Kernel 2: edge MLP via WMMA. Each wave handles 32 edges per iteration:
//   E(32x192) @ We1(192x64)  ->  48x v_wmma_f32_16x16x32_f16
// with two M-tiles sharing every B fragment (2 WMMAs per B load pair).
// ---------------------------------------------------------------------------
__global__ void edge_mlp_kernel(const long long* __restrict__ pairs,
                                const f16* __restrict__ H,
                                const f16* __restrict__ WeT_g,   // [64][192] f16
                                const float* __restrict__ be1,
                                const float* __restrict__ We2,
                                const float* __restrict__ be2p,
                                float* __restrict__ out, int nEdges) {
    __shared__ __align__(16) f16 WeT[HID][KPAD];            // We1^T [n][k], f16
    __shared__ __align__(16) f16 Et[WAVES][MTILE][KPAD];    // per-wave edge tile
    __shared__ float sBe1[HID];
    __shared__ float sW2[HID];

    const int tid = threadIdx.x;

    // Stage WeT_g (64x192 f16) into padded LDS: 1536 v8h chunks / 64 threads
    for (int c = tid; c < HID * (E3 / 8); c += blockDim.x) {
        const int row = c / (E3 / 8);
        const int off = (c % (E3 / 8)) * 8;
        *(v8h*)&WeT[row][off] = *(const v8h*)(WeT_g + row * E3 + off);
    }
    if (tid < HID) { sBe1[tid] = be1[tid]; sW2[tid] = We2[tid]; }
    __syncthreads();

    const int wave = tid >> 5;      // 0..WAVES-1
    const int ln   = tid & 31;
    const int m    = ln & 15;       // A row within 16-tile / B column
    const int part = ln >> 4;       // K half selector for A/B fragments
    const float be2 = be2p[0];

    // Per-lane epilogue constants: this lane owns column N = m + 16*nt of C
    float be1v[4], w2v[4];
#pragma unroll
    for (int t = 0; t < 4; ++t) { be1v[t] = sBe1[m + 16 * t]; w2v[t] = sW2[m + 16 * t]; }

    const int numTiles   = nEdges / MTILE;
    const int totalWaves = gridDim.x * WAVES;

    union AV { v16h v; v8h h[2]; };

    for (int tile = blockIdx.x * WAVES + wave; tile < numTiles; tile += totalWaves) {
        const int e0 = tile * MTILE;

        // ---- Gather: lane ln handles edge e0+ln: copy hu and hv (64 f16 each)
        const long long u = pairs[2 * (long long)(e0 + ln) + 0];
        const long long v = pairs[2 * (long long)(e0 + ln) + 1];
        {
            const v8h* su = (const v8h*)(H + (size_t)u * HID);
            const v8h* sv = (const v8h*)(H + (size_t)v * HID);
            v8h* du = (v8h*)&Et[wave][ln][0];
            v8h* dv = (v8h*)&Et[wave][ln][HID];
#pragma unroll
            for (int i = 0; i < 8; ++i) { du[i] = su[i]; dv[i] = sv[i]; }
        }
        asm volatile("s_wait_dscnt 0" ::: "memory");   // wave-local LDS fence

        // ---- |hu - hv| section: lane covers its own row, all 64 cols
#pragma unroll
        for (int c8 = 0; c8 < 8; ++c8) {
            const int c = c8 * 8;
            v8h hu = *(const v8h*)&Et[wave][ln][c];
            v8h hv = *(const v8h*)&Et[wave][ln][HID + c];
            union { v8h h; v8u uu; } d;
            d.h = hu - hv;
            d.uu = d.uu & (unsigned short)0x7FFF;      // clear sign -> abs
            *(v8h*)&Et[wave][ln][2 * HID + c] = d.h;
        }
        asm volatile("s_wait_dscnt 0" ::: "memory");

        // ---- GEMM: kc outer; A0 (rows 0-15) and A1 (rows 16-31) share B frags
        v8f acc0[4], acc1[4];
#pragma unroll
        for (int nt = 0; nt < 4; ++nt) { acc0[nt] = (v8f)(0.0f); acc1[nt] = (v8f)(0.0f); }

        const int krow = part * 8;
#pragma unroll
        for (int kc = 0; kc < 6; ++kc) {
            // A fragments (ISA 16-bit 16x32 layout): lane m holds row M=m;
            // h[0]=K k0+part*8..+7, h[1]=K k0+16+part*8..+7
            AV a0, a1;
            {
                const f16* b0 = &Et[wave][m][kc * 32 + krow];
                const f16* b1p = &Et[wave][16 + m][kc * 32 + krow];
                a0.h[0] = *(const v8h*)(b0);
                a0.h[1] = *(const v8h*)(b0 + 16);
                a1.h[0] = *(const v8h*)(b1p);
                a1.h[1] = *(const v8h*)(b1p + 16);
            }
#pragma unroll
            for (int nt = 0; nt < 4; ++nt) {
                // B frag (32x16): col = m; h[0]=K k0+part*8..+7, h[1]=K k0+16+part*8..+7
                AV b;
                const f16* bb = &WeT[m + nt * 16][kc * 32 + part * 8];
                b.h[0] = *(const v8h*)(bb);
                b.h[1] = *(const v8h*)(bb + 16);
                acc0[nt] = __builtin_amdgcn_wmma_f32_16x16x32_f16(
                    false, a0.v, false, b.v, (short)0, acc0[nt], false, false);
                acc1[nt] = __builtin_amdgcn_wmma_f32_16x16x32_f16(
                    false, a1.v, false, b.v, (short)0, acc1[nt], false, false);
            }
        }

        // ---- Epilogue: relu(C + be1) . We2 per row, reduce over 16-lane halves
        // C layout: VGPR r, lane -> row M = r + 8*part (+16 for acc1), col N = m+16*nt
        float p0[8], p1[8];
#pragma unroll
        for (int r = 0; r < 8; ++r) {
            float s0 = 0.0f, s1 = 0.0f;
#pragma unroll
            for (int nt = 0; nt < 4; ++nt) {
                s0 += fmaxf(acc0[nt][r] + be1v[nt], 0.0f) * w2v[nt];
                s1 += fmaxf(acc1[nt][r] + be1v[nt], 0.0f) * w2v[nt];
            }
            p0[r] = s0; p1[r] = s1;
        }
#pragma unroll
        for (int off = 8; off >= 1; off >>= 1) {
#pragma unroll
            for (int r = 0; r < 8; ++r) {
                p0[r] += __shfl_xor(p0[r], off, 32);
                p1[r] += __shfl_xor(p1[r], off, 32);
            }
        }
        if (m == 0) {
            const int rbase = e0 + part * 8;
#pragma unroll
            for (int r = 0; r < 8; ++r) {
                out[rbase + r]      = p0[r] + be2;
                out[rbase + 16 + r] = p1[r] + be2;
            }
        }
    }
}

// ---------------------------------------------------------------------------
// Scalar tail kernel for edges beyond the last full 32-edge tile (not hit for
// N_EDGES = 2,000,000, but kept for generality).
// ---------------------------------------------------------------------------
__global__ void edge_tail_kernel(const long long* __restrict__ pairs,
                                 const f16* __restrict__ H,
                                 const float* __restrict__ We1,
                                 const float* __restrict__ be1,
                                 const float* __restrict__ We2,
                                 const float* __restrict__ be2p,
                                 float* __restrict__ out, int nEdges, int rem) {
    const int t = threadIdx.x;
    if (t >= rem) return;
    const int e = nEdges - rem + t;
    const long long u = pairs[2 * (long long)e];
    const long long v = pairs[2 * (long long)e + 1];
    float Ev[E3];
    for (int i = 0; i < HID; ++i) {
        const float hu = (float)H[(size_t)u * HID + i];
        const float hv = (float)H[(size_t)v * HID + i];
        Ev[i] = hu; Ev[HID + i] = hv; Ev[2 * HID + i] = fabsf(hu - hv);
    }
    float s = 0.0f;
    for (int n = 0; n < HID; ++n) {
        float a = be1[n];
        for (int k = 0; k < E3; ++k) a = fmaf(Ev[k], We1[k * HID + n], a);
        s += fmaxf(a, 0.0f) * We2[n];
    }
    out[e] = s + be2p[0];
}

// ---------------------------------------------------------------------------
extern "C" void kernel_launch(void* const* d_in, const int* in_sizes, int n_in,
                              void* d_out, int out_size, void* d_ws, size_t ws_size,
                              hipStream_t stream) {
    const float*     X     = (const float*)d_in[0];
    const long long* pairs = (const long long*)d_in[1];
    const float*     W1    = (const float*)d_in[2];
    const float*     b1    = (const float*)d_in[3];
    const float*     W2    = (const float*)d_in[4];
    const float*     b2    = (const float*)d_in[5];
    const float*     We1   = (const float*)d_in[6];
    const float*     be1   = (const float*)d_in[7];
    const float*     We2   = (const float*)d_in[8];
    const float*     be2   = (const float*)d_in[9];
    float* out = (float*)d_out;

    const int nNodes = in_sizes[0] / DIN;
    const int nEdges = in_sizes[1] / 2;

    // Workspace layout: H (nNodes*64 f16, ~12.8 MB, L2-resident) | WeT (64*192 f16)
    f16* H     = (f16*)d_ws;
    f16* WeT_g = H + (size_t)nNodes * HID;

    // Phase 0: one-off We1 transpose to f16
    we1_transpose_kernel<<<(E3 * HID + 255) / 256, 256, 0, stream>>>(We1, WeT_g);

    // Phase 1: node MLP
    node_mlp_kernel<<<(nNodes + 3) / 4, 256, 0, stream>>>(X, W1, b1, W2, b2, H, nNodes);

    // Phase 2: edge MLP (WMMA)
    const int numTiles = nEdges / MTILE;
    int blocks = (numTiles + WAVES - 1) / WAVES;
    if (blocks > 2048) blocks = 2048;
    if (blocks < 1) blocks = 1;
    edge_mlp_kernel<<<blocks, 32 * WAVES, 0, stream>>>(pairs, H, WeT_g, be1, We2, be2,
                                                       out, nEdges);
    const int rem = nEdges % MTILE;
    if (rem)
        edge_tail_kernel<<<1, 32, 0, stream>>>(pairs, H, We1, be1, We2, be2,
                                               out, nEdges, rem);
}
